// AttentionLayer_43731357008171
// MI455X (gfx1250) — compile-verified, matrix operand order checked by reading, not run on hardware
//
#include <hip/hip_runtime.h>

typedef __attribute__((ext_vector_type(16))) __bf16 v16bf;
typedef __attribute__((ext_vector_type(8)))  __bf16 v8bf;
typedef __attribute__((ext_vector_type(8)))  float  v8f;

#define NBATCH  2048
#define NI      16
#define DIM     768
#define HID     64
#define BB      8      // batch elements per workgroup
#define KSTEPS  24     // 768 / 32
#define NTILES  4      // 64 / 16
#define XSTRIDE 776    // padded LDS row stride (bf16 elems); 1552B, 16B aligned, bank-rotating

// Packed weight fragment offset: [p(own/ev)][i][nt][k][lane] -> 16 bf16 (32B) per lane.
__device__ __forceinline__ size_t pk_off(int p, int i, int nt, int k, int lane) {
  return (((((size_t)p * NI + i) * NTILES + nt) * KSTEPS + k) * 32 + lane) * 16;
}

// ---------------------------------------------------------------------------
// Prep: convert W1 (16 x 1536 x 64 f32) into bf16 WMMA-B fragments in d_ws.
// B operand striping mirrors the 16-bit A layout: lane = column n (lane&15),
// halves split K as {0..7,16..23} / {8..15,24..31} within each 32-K step.
// ---------------------------------------------------------------------------
__global__ void prep_pack_kernel(const float* __restrict__ W1, __bf16* __restrict__ pk) {
  int gid = blockIdx.x * blockDim.x + threadIdx.x;
  if (gid >= 2 * NI * NTILES * KSTEPS * 32) return;
  int lane = gid & 31;
  int t = gid >> 5;
  int k  = t % KSTEPS; t /= KSTEPS;
  int nt = t % NTILES; t /= NTILES;
  int i  = t % NI;     t /= NI;
  int p  = t;                                   // 0 = W1_own, 1 = W1_ev
  int n     = nt * 16 + (lane & 15);
  int kbase = k * 32 + (lane >> 4) * 8;
  __bf16* dst = pk + (size_t)gid * 16;
  const float* src = W1 + (size_t)i * (2 * DIM) * HID;
#pragma unroll
  for (int e = 0; e < 16; ++e) {
    int kk = kbase + (e < 8 ? e : e + 8);
    dst[e] = (__bf16)src[(size_t)(p * DIM + kk) * HID + n];
  }
}

// Assemble a 16-bit A fragment from LDS: two contiguous 16B chunks per lane.
__device__ __forceinline__ v16bf load_a_frag(const __bf16* ap) {
  v8bf lo = *(const v8bf*)ap;          // K kbase .. kbase+7
  v8bf hi = *(const v8bf*)(ap + 16);   // K kbase+16 .. kbase+23
  return __builtin_shufflevector(lo, hi, 0,1,2,3,4,5,6,7,8,9,10,11,12,13,14,15);
}

// ---------------------------------------------------------------------------
// Main: one workgroup = BB batch elements, 16 waves, wave w owns i = w.
// All 8 batch accumulators live simultaneously so every B fragment is loaded
// from L2 exactly once per wave.
// ---------------------------------------------------------------------------
__launch_bounds__(512, 1)
__global__ void attn_main_kernel(const float* __restrict__ X,
                                 const __bf16* __restrict__ pk,
                                 const float* __restrict__ b1,
                                 const float* __restrict__ W2,
                                 float* __restrict__ out) {
  __shared__ __bf16 Xs[BB * 16 * XSTRIDE];        // ~199 KB of the 320 KB WGP LDS
  __shared__ float  own_s[NI][BB][HID];           // 32 KB
  __shared__ float  w_s[NI][BB][16];              // 8 KB

  const int tid  = threadIdx.x;
  const int lane = tid & 31;
  const int i    = tid >> 5;        // wave id == i index
  const int half = lane >> 4;
  const int ln   = lane & 15;
  const int kcol = half * 8;        // per-half K base inside a 32-K step
  const size_t bbase = (size_t)blockIdx.x * BB * NI * DIM;

  // Stage X tile (BB*16 rows x 768) into LDS as bf16 (vectorized, coalesced).
  for (int it = tid; it < BB * 16 * (DIM / 4); it += 512) {
    int row = it / (DIM / 4);
    int c4  = (it % (DIM / 4)) * 4;
    const float* s = X + bbase + (size_t)row * DIM + c4;
    __bf16* d = &Xs[row * XSTRIDE + c4];
    d[0] = (__bf16)s[0]; d[1] = (__bf16)s[1];
    d[2] = (__bf16)s[2]; d[3] = (__bf16)s[3];
  }
  __syncthreads();

  // ---- own pass: A rows = x_{b,i} for bb = 0..7 (rows 8..15 dummies, ignored)
  for (int nt = 0; nt < NTILES; ++nt) {
    v8f c = {};
    for (int k = 0; k < KSTEPS; ++k) {
      v16bf b = *(const v16bf*)(pk + pk_off(0, i, nt, k, lane));
      int rr = (ln < BB) ? ln : 0;
      v16bf a = load_a_frag(&Xs[(rr * 16 + i) * XSTRIDE + k * 32 + kcol]);
      c = __builtin_amdgcn_wmma_f32_16x16x32_bf16(false, a, false, b, (short)0, c, false, false);
    }
    if (half == 0) {                 // rows 0..7 (= bb) live in half 0
      int h = nt * 16 + ln;
      float bias = b1[i * HID + h];
#pragma unroll
      for (int v = 0; v < 8; ++v) own_s[i][v][h] = c[v] + bias;
    }
  }
  // own_s / w_s are private to this wave: no barrier needed.

  // ---- ev pass: all 8 b's share each B fragment (single L2 read per frag).
  float psum[BB][8];
#pragma unroll
  for (int q = 0; q < BB; ++q)
#pragma unroll
    for (int v = 0; v < 8; ++v) psum[q][v] = 0.f;

  for (int nt = 0; nt < NTILES; ++nt) {
    v8f c[BB] = {};
    for (int k = 0; k < KSTEPS; ++k) {
      v16bf b = *(const v16bf*)(pk + pk_off(1, i, nt, k, lane));
#pragma unroll
      for (int q = 0; q < BB; ++q) {
        v16bf a = load_a_frag(&Xs[(q * 16 + ln) * XSTRIDE + k * 32 + kcol]);
        c[q] = __builtin_amdgcn_wmma_f32_16x16x32_bf16(false, a, false, b, (short)0, c[q], false, false);
      }
    }
    int h = nt * 16 + ln;
    float w2v = W2[i * HID + h];
#pragma unroll
    for (int q = 0; q < BB; ++q) {
      float ow = own_s[i][q][h];
#pragma unroll
      for (int v = 0; v < 8; ++v) {
        float val = c[q][v] + ow;           // + b1 already folded into own_s
        val = fmaxf(val, 0.f);              // ReLU
        psum[q][v] += val * w2v;            // partial score over column h
      }
    }
  }

  // Reduce over columns, softmax over j (b2 dropped: shift-invariant).
#pragma unroll
  for (int q = 0; q < BB; ++q) {
#pragma unroll
    for (int m = 1; m < 16; m <<= 1)
#pragma unroll
      for (int v = 0; v < 8; ++v)
        psum[q][v] += __shfl_xor(psum[q][v], m, 32);
    // half 0 now holds scores j=0..7 in psum[q][v], half 1 holds j=8..15.
    float mv = psum[q][0];
#pragma unroll
    for (int v = 1; v < 8; ++v) mv = fmaxf(mv, psum[q][v]);
    mv = fmaxf(mv, __shfl_xor(mv, 16, 32));
    float ev[8], ssum = 0.f;
#pragma unroll
    for (int v = 0; v < 8; ++v) { ev[v] = __expf(psum[q][v] - mv); ssum += ev[v]; }
    ssum += __shfl_xor(ssum, 16, 32);
    float inv = 1.f / ssum;
    if (ln == 0)
#pragma unroll
      for (int v = 0; v < 8; ++v) w_s[i][q][v + 8 * half] = ev[v] * inv;
  }

  // ---- output mix: out[b,i,:] = sum_j w[j] * x_{b,j}  (f32 X from global)
  const float* xb = X + bbase;
  for (int bb = 0; bb < BB; ++bb) {
    float wj[16];
#pragma unroll
    for (int j = 0; j < 16; ++j) wj[j] = w_s[i][bb][j];
    float* op = out + (((size_t)(blockIdx.x * BB + bb) * NI + i) * DIM);
    for (int t = 0; t < DIM / 32; ++t) {
      int d = t * 32 + lane;
      float acc = 0.f;
#pragma unroll
      for (int j = 0; j < 16; ++j)
        acc += wj[j] * xb[(size_t)(bb * 16 + j) * DIM + d];
      op[d] = acc;
    }
  }
}

extern "C" void kernel_launch(void* const* d_in, const int* in_sizes, int n_in,
                              void* d_out, int out_size, void* d_ws, size_t ws_size,
                              hipStream_t stream) {
  const float* X  = (const float*)d_in[0];   // (2048, 16, 768)
  const float* W1 = (const float*)d_in[1];   // (16, 1536, 64)
  const float* b1 = (const float*)d_in[2];   // (16, 64)
  const float* W2 = (const float*)d_in[3];   // (16, 64)
  // d_in[4] = b2: dropped, softmax is shift-invariant.
  __bf16* pk = (__bf16*)d_ws;                // 3 MB packed bf16 W1 fragments
  float* out = (float*)d_out;

  int pk_work = 2 * NI * NTILES * KSTEPS * 32;
  prep_pack_kernel<<<(pk_work + 255) / 256, 256, 0, stream>>>(W1, pk);
  attn_main_kernel<<<NBATCH / BB, 512, 0, stream>>>(X, pk, b1, W2, out);
}
